// MetaBind_MultiEdges_45492293599381
// MI455X (gfx1250) — matrix operand order repaired and checked by reading, not compiled, not run on hardware
//
#include <hip/hip_runtime.h>
#include <hip/hip_bf16.h>

// ---------- types ----------
typedef __attribute__((ext_vector_type(16))) _Float16 v16h;
typedef __attribute__((ext_vector_type(8)))  _Float16 v8h;
typedef __attribute__((ext_vector_type(8)))  float    v8f;

#define WMMA_F16(a, b, c) \
  __builtin_amdgcn_wmma_f32_16x16x32_f16(false, (a), false, (b), (short)0, (c), false, false)

// A-operand (16x32 f16, M x K) per-lane fetch from an LDS row:
// lanes 0-15 hold row M=lane, k in {0..15}; lanes 16-31 hold row M=lane-16,
// k in {16..31}. Two contiguous 16B runs per lane.
__device__ __forceinline__ v16h load_a16(const _Float16* rowBase, int lane, int kc) {
  const int kb0 = (lane < 16) ? 0 : 16;
  const int kb1 = (lane < 16) ? 8 : 24;
  v8h lo = *(const v8h*)(rowBase + kc * 32 + kb0);
  v8h hi = *(const v8h*)(rowBase + kc * 32 + kb1);
  return __builtin_shufflevector(lo, hi, 0,1,2,3,4,5,6,7,8,9,10,11,12,13,14,15);
}

// B-operand tiles pre-swizzled in global memory: tile t = kc*ntiles + nt,
// 32 lanes x 16 halves contiguous; lane holds column n = nt*16 + (lane&15),
// k = kc*32 + (lane<16 ? 0 : 16) + h.
__device__ __forceinline__ v16h load_b16(const _Float16* w, int tile, int lane) {
  return *(const v16h*)(w + (size_t)tile * 512 + lane * 16);
}

__device__ __forceinline__ unsigned enc_f32(float f) {
  unsigned u = __float_as_uint(f);
  return (u & 0x80000000u) ? ~u : (u | 0x80000000u);
}
__device__ __forceinline__ float dec_f32(unsigned e) {
  unsigned u = (e & 0x80000000u) ? (e ^ 0x80000000u) : ~e;
  return __uint_as_float(u);
}

// ---------- zero workspace ----------
__global__ void zero_kernel(float* __restrict__ p, long n) {
  long i = (long)blockIdx.x * blockDim.x + threadIdx.x;
  long stride = (long)gridDim.x * blockDim.x;
  for (; i < n; i += stride) p[i] = 0.0f;
}

// ---------- fold BN into linear, cast f16, swizzle to WMMA-B layout ----------
__global__ void prep_kernel(const float* __restrict__ W, const float* __restrict__ b,
                            const float* __restrict__ g, const float* __restrict__ beta,
                            const float* __restrict__ m, const float* __restrict__ vv,
                            int Ksrc, int Kpad, int Nout,
                            _Float16* __restrict__ wdst, float* __restrict__ bdst) {
  const int ntiles = Nout / 16;
  long total = (long)(Kpad / 32) * ntiles * 512;
  long t = (long)blockIdx.x * blockDim.x + threadIdx.x;
  long stride = (long)gridDim.x * blockDim.x;
  for (long i = t; i < total; i += stride) {
    int tile = (int)(i / 512);
    int rem  = (int)(i % 512);
    int lane = rem / 16;
    int h    = rem % 16;
    int kc = tile / ntiles, nt = tile % ntiles;
    int n = nt * 16 + (lane & 15);
    int k = kc * 32 + ((lane < 16) ? 0 : 16) + h;
    float w = (k < Ksrc) ? W[(long)n * Ksrc + k] : 0.0f;
    float s = g[n] * rsqrtf(vv[n] + 1e-5f);
    wdst[i] = (_Float16)(w * s);
  }
  if (t < Nout) {
    int n = (int)t;
    float s = g[n] * rsqrtf(vv[n] + 1e-5f);
    bdst[n] = s * (b[n] - m[n]) + beta[n];
  }
}

// ---------- node counts per graph ----------
__global__ void count_kernel(const int* __restrict__ batch, float* __restrict__ cnt, int Nn) {
  int i = blockIdx.x * blockDim.x + threadIdx.x;
  int stride = gridDim.x * blockDim.x;
  for (; i < Nn; i += stride) atomicAdd(&cnt[batch[i]], 1.0f);
}

// ---------- edge model: gather -> (176->128 relu) -> (128->128) -> scatter-add ----------
// 128 threads = 4 waves; each wave owns 32 edges (two 16-row A tiles so every
// B weight tile feeds two WMMAs). N is processed in two 4-tile passes to keep
// VGPR pressure low; per K-chunk the 4 B tiles are loaded as one clause so the
// scheduler can overlap loads with WMMAs via partial loadcnt waits.
__global__ __launch_bounds__(128) void edge_kernel(
    const float* __restrict__ x, const float* __restrict__ u,
    const int* __restrict__ batch, const int* __restrict__ ei,
    const float* __restrict__ ea,
    const _Float16* __restrict__ w1, const float* __restrict__ b1,
    const _Float16* __restrict__ w2, const float* __restrict__ b2,
    float* __restrict__ agg, long Ecnt) {
  const int PITCH = 200;  // halves per row (>=192, 16B aligned, bank-spread)
  const int HP = 136;     // hidden pitch
  __shared__ _Float16 smem[4][32 * 200];
  const int wave = threadIdx.x >> 5;
  const int lane = threadIdx.x & 31;
  _Float16* S = smem[wave];
  const long eBase = (long)blockIdx.x * 128 + wave * 32;

  // gather + concat + cast (K padded 176 -> 192), 32 rows
  for (int idx = lane; idx < 32 * 192; idx += 32) {
    int r = idx / 192, c = idx % 192;
    long e = eBase + r; if (e >= Ecnt) e = Ecnt - 1;
    float val = 0.0f;
    if (c < 176) {
      int src = ei[e];
      if (c < 64)        val = x[(long)src * 64 + c];
      else if (c < 128)  { int dst = ei[Ecnt + e]; val = x[(long)dst * 64 + (c - 64)]; }
      else if (c < 144)  val = ea[e * 16 + (c - 128)];
      else               val = u[(long)batch[src] * 32 + (c - 144)];
    }
    S[r * PITCH + c] = (_Float16)val;
  }
  __syncthreads();

  const _Float16* rowA0 = S + (lane & 15) * PITCH;
  const _Float16* rowA1 = S + (16 + (lane & 15)) * PITCH;

  // layer 1: K=192 (6 chunks) x N=128, two passes of 4 N-tiles, M=32
  for (int half = 0; half < 2; ++half) {
    v8f acc0[4] = {}, acc1[4] = {};
    for (int kc = 0; kc < 6; ++kc) {
      v16h a0 = load_a16(rowA0, lane, kc);
      v16h a1 = load_a16(rowA1, lane, kc);
      v16h b[4];
#pragma unroll
      for (int j = 0; j < 4; ++j) b[j] = load_b16(w1, kc * 8 + half * 4 + j, lane);
#pragma unroll
      for (int j = 0; j < 4; ++j) {
        acc0[j] = WMMA_F16(a0, b[j], acc0[j]);
        acc1[j] = WMMA_F16(a1, b[j], acc1[j]);
      }
    }
    // bias + ReLU -> f16 hidden into LDS (wave-private, from S base, overlapping staging rows)
#pragma unroll
    for (int j = 0; j < 4; ++j) {
      int col = (half * 4 + j) * 16 + (lane & 15);
      float bb = b1[col];
#pragma unroll
      for (int i = 0; i < 8; ++i) {
        int row = i + ((lane >= 16) ? 8 : 0);
        float v0 = acc0[j][i] + bb; v0 = v0 > 0.0f ? v0 : 0.0f;
        float v1 = acc1[j][i] + bb; v1 = v1 > 0.0f ? v1 : 0.0f;
        S[row * HP + col] = (_Float16)v0;
        S[(row + 16) * HP + col] = (_Float16)v1;
      }
    }
  }
  __syncthreads();

  // layer 2: K=128 (4 chunks) x N=128, two passes of 4 N-tiles, M=32
  const _Float16* rowH0 = S + (lane & 15) * HP;
  const _Float16* rowH1 = S + (16 + (lane & 15)) * HP;
  for (int half = 0; half < 2; ++half) {
    v8f acc0[4] = {}, acc1[4] = {};
    for (int kc = 0; kc < 4; ++kc) {
      v16h a0 = load_a16(rowH0, lane, kc);
      v16h a1 = load_a16(rowH1, lane, kc);
      v16h b[4];
#pragma unroll
      for (int j = 0; j < 4; ++j) b[j] = load_b16(w2, kc * 8 + half * 4 + j, lane);
#pragma unroll
      for (int j = 0; j < 4; ++j) {
        acc0[j] = WMMA_F16(a0, b[j], acc0[j]);
        acc1[j] = WMMA_F16(a1, b[j], acc1[j]);
      }
    }
    // bias + scatter-add to destination nodes
#pragma unroll
    for (int j = 0; j < 4; ++j) {
      int col = (half * 4 + j) * 16 + (lane & 15);
      float bb = b2[col];
#pragma unroll
      for (int i = 0; i < 8; ++i) {
        int row = i + ((lane >= 16) ? 8 : 0);
        long e0 = eBase + row;
        long e1 = eBase + row + 16;
        if (e0 < Ecnt) {
          int dst = ei[Ecnt + e0];
          atomicAdd(&agg[(long)dst * 128 + col], acc0[j][i] + bb);
        }
        if (e1 < Ecnt) {
          int dst = ei[Ecnt + e1];
          atomicAdd(&agg[(long)dst * 128 + col], acc1[j][i] + bb);
        }
      }
    }
  }
}

// ---------- node model: gather -> (352->128 relu) -> (128->128) -> fused max/mean pool ----------
// 128 threads = 4 waves; each wave owns 16 nodes. xn never hits HBM.
__global__ __launch_bounds__(128) void node_kernel(
    const float* __restrict__ x, const float* __restrict__ u,
    const int* __restrict__ batch,
    const float* __restrict__ agg1, const float* __restrict__ agg2,
    const _Float16* __restrict__ w1, const float* __restrict__ b1,
    const _Float16* __restrict__ w2, const float* __restrict__ b2,
    float* __restrict__ sum, unsigned* __restrict__ mxe, int Nn) {
  const int PITCH = 360;  // >=352 halves, 16B aligned
  const int HP = 136;
  __shared__ _Float16 smem[4][16 * 360];
  const int wave = threadIdx.x >> 5;
  const int lane = threadIdx.x & 31;
  _Float16* S = smem[wave];
  const long nBase = (long)blockIdx.x * 64 + wave * 16;

  for (int idx = lane; idx < 16 * 352; idx += 32) {
    int r = idx / 352, c = idx % 352;
    long nd = nBase + r; if (nd >= Nn) nd = Nn - 1;
    float val;
    if (c < 64)        val = x[nd * 64 + c];
    else if (c < 192)  val = agg1[nd * 128 + (c - 64)];
    else if (c < 320)  val = agg2[nd * 128 + (c - 192)];
    else               val = u[(long)batch[nd] * 32 + (c - 320)];
    S[r * PITCH + c] = (_Float16)val;
  }
  __syncthreads();

  const _Float16* rowA = S + (lane & 15) * PITCH;
  // layer 1: K=352 (11 chunks) x N=128, two passes of 4 N-tiles
  for (int half = 0; half < 2; ++half) {
    v8f acc[4] = {};
    for (int kc = 0; kc < 11; ++kc) {
      v16h a = load_a16(rowA, lane, kc);
      v16h b[4];
#pragma unroll
      for (int j = 0; j < 4; ++j) b[j] = load_b16(w1, kc * 8 + half * 4 + j, lane);
#pragma unroll
      for (int j = 0; j < 4; ++j) acc[j] = WMMA_F16(a, b[j], acc[j]);
    }
#pragma unroll
    for (int j = 0; j < 4; ++j) {
      int col = (half * 4 + j) * 16 + (lane & 15);
      float bb = b1[col];
#pragma unroll
      for (int i = 0; i < 8; ++i) {
        float v = acc[j][i] + bb;
        v = v > 0.0f ? v : 0.0f;
        int row = i + ((lane >= 16) ? 8 : 0);
        S[row * HP + col] = (_Float16)v;
      }
    }
  }
  __syncthreads();

  const _Float16* rowH = S + (lane & 15) * HP;
  // layer 2: K=128 (4 chunks) x N=128, two passes of 4 N-tiles, fused pooling
  for (int half = 0; half < 2; ++half) {
    v8f acc[4] = {};
    for (int kc = 0; kc < 4; ++kc) {
      v16h a = load_a16(rowH, lane, kc);
      v16h b[4];
#pragma unroll
      for (int j = 0; j < 4; ++j) b[j] = load_b16(w2, kc * 8 + half * 4 + j, lane);
#pragma unroll
      for (int j = 0; j < 4; ++j) acc[j] = WMMA_F16(a, b[j], acc[j]);
    }
#pragma unroll
    for (int j = 0; j < 4; ++j) {
      int col = (half * 4 + j) * 16 + (lane & 15);
      float bb = b2[col];
#pragma unroll
      for (int i = 0; i < 8; ++i) {
        long nd = nBase + i + ((lane >= 16) ? 8 : 0);
        if (nd < Nn) {
          float v = acc[j][i] + bb;
          int g = batch[nd];
          atomicAdd(&sum[(long)g * 128 + col], v);
          atomicMax(&mxe[(long)g * 128 + col], enc_f32(v));
        }
      }
    }
  }
}

// ---------- global model: one wave, M=16 graphs = one WMMA tile ----------
__global__ __launch_bounds__(32) void global_kernel(
    const float* __restrict__ u, const float* __restrict__ sum,
    const unsigned* __restrict__ mxe, const float* __restrict__ cnt,
    const _Float16* __restrict__ w1, const float* __restrict__ b1,
    const _Float16* __restrict__ w2, const float* __restrict__ b2,
    float* __restrict__ out) {
  const int KP = 296, HP = 40;
  __shared__ _Float16 zg[16 * 296];
  __shared__ _Float16 hid[16 * 40];
  const int lane = threadIdx.x;

  for (int idx = lane; idx < 16 * 288; idx += 32) {
    int r = idx / 288, c = idx % 288;
    float cr = cnt[r];
    float v;
    if (c < 32) {
      v = u[r * 32 + c];
    } else if (c < 160) {
      v = dec_f32(mxe[r * 128 + (c - 32)]);
      if (!(cr > 0.0f)) v = 0.0f;
    } else {
      v = sum[r * 128 + (c - 160)] / fmaxf(cr, 1.0f);
    }
    zg[r * KP + c] = (_Float16)v;
  }
  __syncthreads();

  v8f acc[2] = {};
  const _Float16* rowA = zg + (lane & 15) * KP;
  for (int kc = 0; kc < 9; ++kc) {
    v16h a = load_a16(rowA, lane, kc);
    v16h b[2];
#pragma unroll
    for (int nt = 0; nt < 2; ++nt) b[nt] = load_b16(w1, kc * 2 + nt, lane);
#pragma unroll
    for (int nt = 0; nt < 2; ++nt) acc[nt] = WMMA_F16(a, b[nt], acc[nt]);
  }
#pragma unroll
  for (int nt = 0; nt < 2; ++nt) {
    int col = nt * 16 + (lane & 15);
    float bb = b1[col];
#pragma unroll
    for (int i = 0; i < 8; ++i) {
      float v = acc[nt][i] + bb;
      v = v > 0.0f ? v : 0.0f;
      int row = i + ((lane >= 16) ? 8 : 0);
      hid[row * HP + col] = (_Float16)v;
    }
  }
  __syncthreads();

  v8f acc2[2] = {};
  {
    v16h a = load_a16(hid + (lane & 15) * HP, lane, 0);
#pragma unroll
    for (int nt = 0; nt < 2; ++nt) {
      v16h b = load_b16(w2, nt, lane);
      acc2[nt] = WMMA_F16(a, b, acc2[nt]);
    }
  }
#pragma unroll
  for (int nt = 0; nt < 2; ++nt) {
    int col = nt * 16 + (lane & 15);
    float bb = b2[col];
#pragma unroll
    for (int i = 0; i < 8; ++i) {
      int row = i + ((lane >= 16) ? 8 : 0);
      out[row * 32 + col] = acc2[nt][i] + bb;
    }
  }
}

// ---------- host ----------
extern "C" void kernel_launch(void* const* d_in, const int* in_sizes, int n_in,
                              void* d_out, int out_size, void* d_ws, size_t ws_size,
                              hipStream_t stream) {
  const float* x    = (const float*)d_in[0];
  const int*   ei1  = (const int*)d_in[1];
  const float* ea1  = (const float*)d_in[2];
  const int*   ei2  = (const int*)d_in[3];
  const float* ea2  = (const float*)d_in[4];
  const float* u    = (const float*)d_in[5];
  const int*   batch= (const int*)d_in[6];
  // params flattened in dict order: e1{W,b}, ebn1{g,b,m,v}, e2, ebn2, n1, nbn1, n2, nbn2, g1, gbn1, g2, gbn2
  const float* P[36];
  for (int i = 0; i < 36 && (7 + i) < n_in; ++i) P[i] = (const float*)d_in[7 + i];

  const long Nn = in_sizes[0] / 64;   // 50000
  const long Ec = in_sizes[2] / 16;   // 500000
  const long Bg = in_sizes[5] / 32;   // 16

  // workspace layout (256B aligned segments)
  char* base = (char*)d_ws;
  size_t off = 0;
  auto alloc = [&](size_t bytes) { size_t o = off; off += (bytes + 255) & ~(size_t)255; return o; };
  size_t o_we1 = alloc(6 * 8 * 512 * 2);
  size_t o_we2 = alloc(4 * 8 * 512 * 2);
  size_t o_wn1 = alloc(11 * 8 * 512 * 2);
  size_t o_wn2 = alloc(4 * 8 * 512 * 2);
  size_t o_wg1 = alloc(9 * 2 * 512 * 2);
  size_t o_wg2 = alloc(1 * 2 * 512 * 2);
  size_t o_be1 = alloc(128 * 4);
  size_t o_be2 = alloc(128 * 4);
  size_t o_bn1 = alloc(128 * 4);
  size_t o_bn2 = alloc(128 * 4);
  size_t o_bg1 = alloc(32 * 4);
  size_t o_bg2 = alloc(32 * 4);
  size_t o_agg1 = alloc((size_t)Nn * 128 * 4);
  size_t o_agg2 = alloc((size_t)Nn * 128 * 4);
  size_t o_sum  = alloc((size_t)Bg * 128 * 4);
  size_t o_mxe  = alloc((size_t)Bg * 128 * 4);
  size_t o_cnt  = alloc((size_t)Bg * 4);
  size_t o_end  = off;

  _Float16* we1 = (_Float16*)(base + o_we1);
  _Float16* we2 = (_Float16*)(base + o_we2);
  _Float16* wn1 = (_Float16*)(base + o_wn1);
  _Float16* wn2 = (_Float16*)(base + o_wn2);
  _Float16* wg1 = (_Float16*)(base + o_wg1);
  _Float16* wg2 = (_Float16*)(base + o_wg2);
  float* be1 = (float*)(base + o_be1);
  float* be2 = (float*)(base + o_be2);
  float* bn1 = (float*)(base + o_bn1);
  float* bn2 = (float*)(base + o_bn2);
  float* bg1 = (float*)(base + o_bg1);
  float* bg2 = (float*)(base + o_bg2);
  float* agg1 = (float*)(base + o_agg1);
  float* agg2 = (float*)(base + o_agg2);
  float* sum  = (float*)(base + o_sum);
  unsigned* mxe = (unsigned*)(base + o_mxe);
  float* cnt  = (float*)(base + o_cnt);

  // 1) zero the accumulation region (agg1..cnt is contiguous from o_agg1)
  long zcount = (long)((o_end - o_agg1) / 4);
  zero_kernel<<<1024, 256, 0, stream>>>((float*)(base + o_agg1), zcount);

  // 2) fold BN + cast + swizzle all six layers
  auto prep = [&](const float* W, const float* b, const float* const* bn,
                  int Ksrc, int Kpad, int Nout, _Float16* wd, float* bd) {
    long total = (long)(Kpad / 32) * (Nout / 16) * 512;
    int blocks = (int)((total + 255) / 256);
    prep_kernel<<<blocks, 256, 0, stream>>>(W, b, bn[0], bn[1], bn[2], bn[3],
                                            Ksrc, Kpad, Nout, wd, bd);
  };
  const float* ebn1[4] = {P[2], P[3], P[4], P[5]};
  const float* ebn2[4] = {P[8], P[9], P[10], P[11]};
  const float* nbn1[4] = {P[14], P[15], P[16], P[17]};
  const float* nbn2[4] = {P[20], P[21], P[22], P[23]};
  const float* gbn1[4] = {P[26], P[27], P[28], P[29]};
  const float* gbn2[4] = {P[32], P[33], P[34], P[35]};
  prep(P[0],  P[1],  ebn1, 176, 192, 128, we1, be1);
  prep(P[6],  P[7],  ebn2, 128, 128, 128, we2, be2);
  prep(P[12], P[13], nbn1, 352, 352, 128, wn1, bn1);
  prep(P[18], P[19], nbn2, 128, 128, 128, wn2, bn2);
  prep(P[24], P[25], gbn1, 288, 288, 32,  wg1, bg1);
  prep(P[30], P[31], gbn2, 32,  32,  32,  wg2, bg2);

  // 3) edge models (both edge types) -> scatter-add into agg1/agg2
  int eblocks = (int)((Ec + 127) / 128);
  edge_kernel<<<eblocks, 128, 0, stream>>>(x, u, batch, ei1, ea1, we1, be1, we2, be2, agg1, Ec);
  edge_kernel<<<eblocks, 128, 0, stream>>>(x, u, batch, ei2, ea2, we1, be1, we2, be2, agg2, Ec);

  // 4) node counts per graph
  count_kernel<<<256, 256, 0, stream>>>(batch, cnt, (int)Nn);

  // 5) node model with fused max/mean pooling
  int nblocks = (int)((Nn + 63) / 64);
  node_kernel<<<nblocks, 128, 0, stream>>>(x, u, batch, agg1, agg2, wn1, bn1, wn2, bn2,
                                           sum, mxe, (int)Nn);

  // 6) global model -> d_out [16,32]
  global_kernel<<<1, 32, 0, stream>>>(u, sum, mxe, cnt, wg1, bg1, wg2, bg2, (float*)d_out);
}
